// SparseKnowledgeAttention_85830626443699
// MI455X (gfx1250) — compile-verified, matrix operand order checked by reading, not run on hardware
//
#include <hip/hip_runtime.h>
#include <hip/hip_bf16.h>
#include <stdint.h>

#define D_MODEL 1024
#define N_HEADS 16
#define HEAD_DIM 64
#define TOP_K 32
#define B_SZ 2
#define S_LEN 2048
#define N_KE 2048

typedef __attribute__((ext_vector_type(16))) __bf16 v16bf;
typedef __attribute__((ext_vector_type(8)))  float  v8f;

union BFrag {
  v16bf v;
  unsigned short u[16];
};

// fp32 -> bf16 round-to-nearest-even
__device__ __forceinline__ unsigned short f2bf(float f) {
  unsigned int u = __builtin_bit_cast(unsigned int, f);
  unsigned int r = u + 0x7FFFu + ((u >> 16) & 1u);
  return (unsigned short)(r >> 16);
}

// One coalesced async 16B-per-lane burst: 32 lanes x 16B = 512B -> LDS.
// GVS addressing: saddr = 64-bit base, vaddr = signed 32-bit byte offset.
__device__ __forceinline__ void async_copy_b128(uint32_t lds_byte_addr,
                                                uint32_t gbl_byte_off,
                                                const float* base) {
  asm volatile("global_load_async_to_lds_b128 %0, %1, %2"
               :
               : "v"(lds_byte_addr), "v"(gbl_byte_off), "s"(base)
               : "memory");
}
__device__ __forceinline__ void wait_asynccnt0() {
  asm volatile("s_wait_asynccnt 0x0" ::: "memory");
}
__device__ __forceinline__ void wait_dscnt0() {
  asm volatile("s_wait_dscnt 0x0" ::: "memory");
}

// -----------------------------------------------------------------------------
// NT GEMM: C[M, Nout] = A[M, K] * W[Nout, K]^T + bias.
// Each wave owns a 16x64 C strip: one A fragment feeds 4 WMMAs per k-step.
// A (16x32 f32) and W (64x32 f32) k-slabs are staged into per-wave LDS tiles
// with async global->LDS b128 bursts, then converted to bf16 fragments.
// OUT_BF16 = 1: store bf16 (Q/K); 0: store fp32 (V / final output).
// -----------------------------------------------------------------------------
template <int OUT_BF16>
__global__ void wmma_gemm_nt(const float* __restrict__ A,
                             const float* __restrict__ W,
                             const float* __restrict__ bias,
                             void* __restrict__ out,
                             int M, int K, int Nout) {
  extern __shared__ float smem[];  // per-wave: 16*32 (A) + 64*32 (W) floats
  const int lane = threadIdx.x & 31;
  const int wv   = threadIdx.x >> 5;
  float* aTile = smem + wv * (16 * 32 + 64 * 32);
  float* wTile = aTile + 16 * 32;

  const int wid     = blockIdx.x * (blockDim.x >> 5) + wv;
  const int tilesN4 = Nout >> 6;          // 64-wide column groups
  const int tm  = wid / tilesN4;
  const int tn4 = wid % tilesN4;
  if (tm * 16 >= M) return;               // wave-uniform

  const uint32_t aLds = (uint32_t)(size_t)aTile;
  const uint32_t wLds = (uint32_t)(size_t)wTile;
  const int r16 = lane & 15;
  const int kb  = (lane >> 4) << 3;       // 0 or 8

  // copy lane mapping: 4 rows x 128B per issue
  const int cpRow  = lane >> 3;           // 0..3
  const int cpByte = (lane & 7) << 4;     // 0,16,...,112

  v8f acc[4] = {{}, {}, {}, {}};

  for (int k0 = 0; k0 < K; k0 += 32) {
    wait_dscnt0();  // prior iteration's LDS reads fully drained (WAR safety)

    // Stage A k-slab: 16 rows x 32 f32
#pragma unroll
    for (int i = 0; i < 4; ++i) {
      int row = i * 4 + cpRow;
      async_copy_b128(aLds + (uint32_t)(row * 128 + cpByte),
                      (uint32_t)(((tm * 16 + row) * K + k0) * 4 + cpByte), A);
    }
    // Stage W k-slab: 64 rows x 32 f32
#pragma unroll
    for (int i = 0; i < 16; ++i) {
      int row = i * 4 + cpRow;
      async_copy_b128(wLds + (uint32_t)(row * 128 + cpByte),
                      (uint32_t)(((tn4 * 64 + row) * K + k0) * 4 + cpByte), W);
    }
    wait_asynccnt0();

    // A fragment (16x32 bf16)
    BFrag a;
#pragma unroll
    for (int h = 0; h < 8; ++h) {
      a.u[h]     = f2bf(aTile[r16 * 32 + kb + h]);
      a.u[h + 8] = f2bf(aTile[r16 * 32 + kb + 16 + h]);
    }
    // 4 B fragments / 4 WMMAs reusing the A fragment
#pragma unroll
    for (int t = 0; t < 4; ++t) {
      BFrag bf;
      const int wr = (t * 16 + r16) * 32;
#pragma unroll
      for (int h = 0; h < 8; ++h) {
        bf.u[h]     = f2bf(wTile[wr + kb + h]);
        bf.u[h + 8] = f2bf(wTile[wr + kb + 16 + h]);
      }
      acc[t] = __builtin_amdgcn_wmma_f32_16x16x32_bf16(false, a.v, false, bf.v,
                                                       (short)0, acc[t],
                                                       false, false);
    }
  }

  // C layout: VGPR r: lanes 0-15 -> M=r, lanes 16-31 -> M=r+8; N = lane&15
  const int mbase = tm * 16 + ((lane >> 4) << 3);
#pragma unroll
  for (int t = 0; t < 4; ++t) {
    const int ncol = tn4 * 64 + t * 16 + r16;
    const float bv = bias[ncol];
#pragma unroll
    for (int r = 0; r < 8; ++r) {
      float val = acc[t][r] + bv;
      size_t off = (size_t)(mbase + r) * (size_t)Nout + (size_t)ncol;
      if (OUT_BF16) ((unsigned short*)out)[off] = f2bf(val);
      else          ((float*)out)[off] = val;
    }
  }
}

// -----------------------------------------------------------------------------
// Sparse top-k attention core. One 256-thread workgroup (8 waves) handles 16
// query rows of one (batch, head). Waves split the 2048 keys for the WMMA
// score phase into a shared 16x2048 fp32 LDS stripe (128 KB); then each wave
// privately owns 2 query rows for top-32 / softmax / weighted V-gather.
// Top-k winners live in registers: lane `it` holds winner #it (TOP_K == 32).
// -----------------------------------------------------------------------------
__global__ void sparse_topk_attn(const unsigned short* __restrict__ Qb,
                                 const unsigned short* __restrict__ Kb,
                                 const float* __restrict__ Vf,
                                 float* __restrict__ Ctx) {
  extern __shared__ float sc[];  // 16 * N_KE floats
  const int lane = threadIdx.x & 31;
  const int wv   = threadIdx.x >> 5;  // 0..7
  const int stiles = S_LEN / 16;
  int bid = blockIdx.x;
  const int st = bid % stiles;  bid /= stiles;
  const int h  = bid % N_HEADS; bid /= N_HEADS;
  const int b  = bid;
  const int s0 = st * 16;

  const int r16 = lane & 15;
  const int kb  = (lane >> 4) << 3;

  // Q A-fragments (same 16 rows for every wave): two K=32 slabs over hd=64
  const unsigned short* q =
      Qb + ((size_t)b * S_LEN + s0) * D_MODEL + h * HEAD_DIM;
  BFrag a0, a1;
#pragma unroll
  for (int t = 0; t < 8; ++t) {
    a0.u[t]     = q[(size_t)r16 * D_MODEL + kb + t];
    a0.u[t + 8] = q[(size_t)r16 * D_MODEL + kb + 16 + t];
    a1.u[t]     = q[(size_t)r16 * D_MODEL + 32 + kb + t];
    a1.u[t + 8] = q[(size_t)r16 * D_MODEL + 32 + kb + 16 + t];
  }

  // Score phase: wave wv covers key tiles n0 = wv*16, step 128
  const unsigned short* kbm =
      Kb + (size_t)b * N_KE * D_MODEL + h * HEAD_DIM;
  for (int n0 = wv * 16; n0 < N_KE; n0 += 8 * 16) {
    const unsigned short* kt = kbm + (size_t)(n0 + r16) * D_MODEL;
    BFrag b0, b1;
#pragma unroll
    for (int t = 0; t < 8; ++t) {
      b0.u[t]     = kt[kb + t];
      b0.u[t + 8] = kt[kb + 16 + t];
      b1.u[t]     = kt[32 + kb + t];
      b1.u[t + 8] = kt[32 + kb + 16 + t];
    }
    v8f c = {};
    c = __builtin_amdgcn_wmma_f32_16x16x32_bf16(false, a0.v, false, b0.v,
                                                (short)0, c, false, false);
    c = __builtin_amdgcn_wmma_f32_16x16x32_bf16(false, a1.v, false, b1.v,
                                                (short)0, c, false, false);
    const int mb = (lane >> 4) << 3;
#pragma unroll
    for (int r = 0; r < 8; ++r)
      sc[(mb + r) * N_KE + n0 + r16] = c[r] * 0.125f;  // 1/sqrt(64)
  }
  __syncthreads();

  const float* vbase = Vf + (size_t)b * N_KE * D_MODEL + h * HEAD_DIM;

  // Each wave privately owns rows 2*wv and 2*wv+1
#pragma unroll
  for (int rr = 0; rr < 2; ++rr) {
    const int row = wv * 2 + rr;
    float* srow = sc + row * N_KE;

    float myv = -3.0e38f;  // lane `it` will hold winner #it
    int   myi = 0;
    for (int it = 0; it < TOP_K; ++it) {
      float bv = -3.0e38f;
      int   bi = 0x7FFFFFFF;
#pragma unroll 8
      for (int j = 0; j < N_KE / 32; ++j) {
        float v = srow[lane + 32 * j];
        if (v > bv) { bv = v; bi = lane + 32 * j; }
      }
#pragma unroll
      for (int off = 16; off > 0; off >>= 1) {
        float ov = __shfl_xor(bv, off, 32);
        int   oi = __shfl_xor(bi, off, 32);
        if (ov > bv || (ov == bv && oi < bi)) { bv = ov; bi = oi; }
      }
      if (lane == it) { myv = bv; myi = bi; }
      // mask the winner; it is re-read only by this same lane (in-order LDS)
      if (lane == (bi & 31)) srow[bi] = -3.0e38f;
    }

    // softmax over the 32 winners (lane owns winner #lane; #0 is the max)
    const float mx = __shfl(myv, 0, 32);
    const float e  = __expf(myv - mx);
    float ssum = e;
#pragma unroll
    for (int off = 16; off > 0; off >>= 1) ssum += __shfl_xor(ssum, off, 32);
    const float w = e / ssum;

    // context = sum_k w_k * V[idx_k, :]; lane owns dims {lane, lane+32}
    float acc0 = 0.f, acc1 = 0.f;
    for (int kk = 0; kk < TOP_K; ++kk) {
      const float wk  = __shfl(w, kk, 32);
      const int   idx = __shfl(myi, kk, 32);
      const float* vr = vbase + (size_t)idx * D_MODEL;
      acc0 += wk * vr[lane];
      acc1 += wk * vr[lane + 32];
    }
    float* crow =
        Ctx + ((size_t)b * S_LEN + s0 + row) * D_MODEL + h * HEAD_DIM;
    crow[lane]      = acc0;
    crow[lane + 32] = acc1;
  }
}

// -----------------------------------------------------------------------------
extern "C" void kernel_launch(void* const* d_in, const int* in_sizes, int n_in,
                              void* d_out, int out_size, void* d_ws, size_t ws_size,
                              hipStream_t stream) {
  (void)in_sizes; (void)n_in; (void)out_size; (void)ws_size;

  const float* x  = (const float*)d_in[0];
  const float* ke = (const float*)d_in[1];
  const float* Wq = (const float*)d_in[2];
  const float* bq = (const float*)d_in[3];
  const float* Wk = (const float*)d_in[4];
  const float* bk = (const float*)d_in[5];
  const float* Wv = (const float*)d_in[6];
  const float* bv = (const float*)d_in[7];
  const float* Wo = (const float*)d_in[8];
  const float* bo = (const float*)d_in[9];
  float* out = (float*)d_out;

  char* ws = (char*)d_ws;
  unsigned short* Qb = (unsigned short*)(ws);                               // 8 MB
  unsigned short* Kb = (unsigned short*)(ws + (size_t)8  * 1024 * 1024);    // 8 MB
  float* Vf  = (float*)(ws + (size_t)16 * 1024 * 1024);                     // 16 MB
  float* Ctx = (float*)(ws + (size_t)32 * 1024 * 1024);                     // 16 MB

  const int M = B_SZ * S_LEN;                         // 4096 rows
  const int waveTiles = (M / 16) * (D_MODEL / 64);    // 4096 wave-strips
  dim3 gblk(256);                                     // 8 waves per block
  dim3 ggrid(waveTiles / 8);
  size_t gemm_lds = (size_t)8 * (16 * 32 + 64 * 32) * sizeof(float);  // 80 KB

  // Q/K projections -> bf16 workspace; V projection -> f32 workspace
  wmma_gemm_nt<1><<<ggrid, gblk, gemm_lds, stream>>>(x,  Wq, bq, (void*)Qb, M, D_MODEL, D_MODEL);
  wmma_gemm_nt<1><<<ggrid, gblk, gemm_lds, stream>>>(ke, Wk, bk, (void*)Kb, M, D_MODEL, D_MODEL);
  wmma_gemm_nt<0><<<ggrid, gblk, gemm_lds, stream>>>(ke, Wv, bv, (void*)Vf, M, D_MODEL, D_MODEL);

  // top-k attention core: 8 waves per (b, h, 16-row query stripe)
  dim3 agrid(B_SZ * N_HEADS * (S_LEN / 16));
  size_t attn_lds = (size_t)16 * N_KE * sizeof(float);  // 128 KB stripe
  sparse_topk_attn<<<agrid, dim3(256), attn_lds, stream>>>(Qb, Kb, Vf, Ctx);

  // output projection -> d_out (fp32)
  wmma_gemm_nt<0><<<ggrid, gblk, gemm_lds, stream>>>(Ctx, Wo, bo, (void*)out, M, D_MODEL, D_MODEL);
}
